// MGUCell_15401752723863
// MI455X (gfx1250) — matrix operand order repaired and checked by reading, not compiled, not run on hardware
//
#include <hip/hip_runtime.h>

// MGU cell for MI455X (gfx1250, wave32, WMMA + async-to-LDS).
// B=64, T=2048, IN_DIM=UNITS=256.

typedef __attribute__((ext_vector_type(16))) __bf16        v16bf;
typedef __attribute__((ext_vector_type(8)))  float         v8f;
typedef __attribute__((ext_vector_type(4)))  unsigned int  u32x4;

#define BATCH 64
#define SEQ   2048
#define DIM   256
#define ROWS  (BATCH * SEQ)

union Frag { v16bf v; u32x4 u[2]; };

__device__ __forceinline__ unsigned short f2bf(float f) {
  union { float f; unsigned u; } c; c.f = f;
  unsigned r = c.u + 0x7FFFu + ((c.u >> 16) & 1u);   // round-to-nearest-even
  return (unsigned short)(r >> 16);
}
__device__ __forceinline__ float bf2f(unsigned short h) {
  union { unsigned u; float f; } c; c.u = ((unsigned)h) << 16;
  return c.f;
}

__device__ __forceinline__ v8f wmma_bf16(const Frag& a, const Frag& b, v8f c) {
  // D = A(16x32 bf16) x B(32x16 bf16) + C(16x16 f32)
  return __builtin_amdgcn_wmma_f32_16x16x32_bf16(false, a.v, false, b.v,
                                                 (short)0, c, false, false);
}

// gfx1250 async DMA: 16 bytes global -> LDS, tracked by ASYNCcnt.
__device__ __forceinline__ void async_b128(unsigned ldsoff, const void* g) {
  asm volatile("global_load_async_to_lds_b128 %0, %1, off"
               :: "v"(ldsoff), "v"(g) : "memory");
}
__device__ __forceinline__ void wait_async0() {
  asm volatile("s_wait_asynccnt 0x0" ::: "memory");
}

// ---------------------------------------------------------------------------
// Kernel 0: convert the four 256x256 f32 weight matrices to transposed bf16.
// ---------------------------------------------------------------------------
__global__ __launch_bounds__(256) void k_wconv(
    const float* __restrict__ Wf, const float* __restrict__ Uf,
    const float* __restrict__ Ws, const float* __restrict__ Us,
    unsigned short* __restrict__ WfT, unsigned short* __restrict__ UfT,
    unsigned short* __restrict__ WsT, unsigned short* __restrict__ UsT) {
  const int i = blockIdx.x * 256 + threadIdx.x;   // 0 .. 65535
  const int k = i >> 8, n = i & 255;
  const int o = n * 256 + k;
  WfT[o] = f2bf(Wf[i]);
  UfT[o] = f2bf(Uf[i]);
  WsT[o] = f2bf(Ws[i]);
  UsT[o] = f2bf(Us[i]);
}

// ---------------------------------------------------------------------------
// Kernel 1: xf = X @ W_f + b_f ; xs = X @ W_s + b_s   (stored bf16)
// One WG per 64 flattened rows (4 M-tiles) -> each B-fragment load feeds
// 4 WMMAs. 8 waves; wave w owns 32 output columns for both GEMMs.
// ---------------------------------------------------------------------------
__global__ __launch_bounds__(256) void k_xproj(
    const float* __restrict__ x,
    const unsigned short* __restrict__ WfT, const unsigned short* __restrict__ WsT,
    const float* __restrict__ bfv, const float* __restrict__ bsv,
    unsigned short* __restrict__ xf, unsigned short* __restrict__ xs) {
  __shared__ unsigned short aS[64 * 256];         // 32 KB staged A (bf16)
  const int tid = threadIdx.x;
  const long r0 = (long)blockIdx.x * 64;

  const float* xr = x + r0 * 256;
  #pragma unroll
  for (int i = 0; i < 64; ++i) {                  // 16384 elems / 256 threads
    const int idx = tid + i * 256;
    aS[idx] = f2bf(xr[idx]);
  }
  __syncthreads();

  const int wave  = tid >> 5, lane = tid & 31;
  const int laneM = lane & 15;
  const int koffA = (lane < 16) ? 0 : 16;
  const int koffB = (lane < 16) ? 0 : 16;
  const int n0    = wave * 32;
  const int nB0   = n0 + laneM, nB1 = n0 + 16 + laneM;

  v8f accF[4][2] = {};
  v8f accS[4][2] = {};
  #pragma unroll
  for (int kt = 0; kt < 8; ++kt) {
    const int kb = kt * 32;
    Frag b0f, b1f, b0s, b1s;
    b0f.u[0] = *(const u32x4*)&WfT[nB0 * 256 + kb + koffB];
    b0f.u[1] = *(const u32x4*)&WfT[nB0 * 256 + kb + koffB + 8];
    b1f.u[0] = *(const u32x4*)&WfT[nB1 * 256 + kb + koffB];
    b1f.u[1] = *(const u32x4*)&WfT[nB1 * 256 + kb + koffB + 8];
    b0s.u[0] = *(const u32x4*)&WsT[nB0 * 256 + kb + koffB];
    b0s.u[1] = *(const u32x4*)&WsT[nB0 * 256 + kb + koffB + 8];
    b1s.u[0] = *(const u32x4*)&WsT[nB1 * 256 + kb + koffB];
    b1s.u[1] = *(const u32x4*)&WsT[nB1 * 256 + kb + koffB + 8];
    #pragma unroll
    for (int rt = 0; rt < 4; ++rt) {
      Frag a;
      const int abase = (rt * 16 + laneM) * 256 + kb + koffA;
      a.u[0] = *(const u32x4*)&aS[abase];
      a.u[1] = *(const u32x4*)&aS[abase + 8];
      accF[rt][0] = wmma_bf16(a, b0f, accF[rt][0]);
      accF[rt][1] = wmma_bf16(a, b1f, accF[rt][1]);
      accS[rt][0] = wmma_bf16(a, b0s, accS[rt][0]);
      accS[rt][1] = wmma_bf16(a, b1s, accS[rt][1]);
    }
  }

  const float bf0 = bfv[nB0], bf1 = bfv[nB1];
  const float bs0 = bsv[nB0], bs1 = bsv[nB1];
  #pragma unroll
  for (int rt = 0; rt < 4; ++rt) {
    #pragma unroll
    for (int r = 0; r < 8; ++r) {
      const int Ml = (lane < 16) ? r : r + 8;     // C layout: M = r + 8*(lane>=16)
      const long row = r0 + rt * 16 + Ml;
      xf[row * 256 + nB0] = f2bf(accF[rt][0][r] + bf0);
      xf[row * 256 + nB1] = f2bf(accF[rt][1][r] + bf1);
      xs[row * 256 + nB0] = f2bf(accS[rt][0][r] + bs0);
      xs[row * 256 + nB1] = f2bf(accS[rt][1][r] + bs1);
    }
  }
}

// ---------------------------------------------------------------------------
// Kernel 2: recurrent scan. 4 WGs x 512 threads (16 waves); WG owns 16 batch
// rows, wave owns 16 hidden columns. U_f^T / U_s^T fragments resident in
// VGPRs; h in LDS (f32 + bf16 A-copy); xf/xs timestep lines double-buffered
// into LDS with global_load_async_to_lds_b128 overlapped with both GEMMs.
// ---------------------------------------------------------------------------
__global__ __launch_bounds__(512) void k_scan(
    const unsigned short* __restrict__ xf, const unsigned short* __restrict__ xs,
    const unsigned short* __restrict__ UfT, const unsigned short* __restrict__ UsT,
    float* __restrict__ out) {
  __shared__ float          hF[16 * 256];       // h, exact f32
  __shared__ unsigned short hA[16 * 256];       // h, bf16 (WMMA A source)
  __shared__ unsigned short gA[16 * 256];       // f*h, bf16 (WMMA A source)
  __shared__ unsigned short xfL[2][16 * 256];   // staged xf_t (double buffer)
  __shared__ unsigned short xsL[2][16 * 256];   // staged xs_t

  const int tid   = threadIdx.x;
  const int wave  = tid >> 5, lane = tid & 31;
  const int laneM = lane & 15;
  const int koffA = (lane < 16) ? 0 : 16;
  const int koffB = (lane < 16) ? 0 : 16;
  const int nG    = wave * 16 + laneM;          // this lane's hidden column
  const int b0    = blockIdx.x * 16;            // batch rows of this WG

  // Per-thread async-copy chunk: 16 bytes of one staged row.
  const int crow = tid >> 5;                    // 0..15 (staged row)
  const int ccol = (tid & 31) * 8;              // element offset (8 bf16 = 16B)
  const size_t gchunk = (size_t)(b0 + crow) * SEQ * 256 + ccol;   // + t*256
  const unsigned offF[2] = {
      (unsigned)(size_t)&xfL[0][crow * 256 + ccol],
      (unsigned)(size_t)&xfL[1][crow * 256 + ccol] };
  const unsigned offS[2] = {
      (unsigned)(size_t)&xsL[0][crow * 256 + ccol],
      (unsigned)(size_t)&xsL[1][crow * 256 + ccol] };

  // Preload recurrent weight B-fragments into registers (stay resident).
  Frag Bf[8], Bs[8];
  #pragma unroll
  for (int kt = 0; kt < 8; ++kt) {
    const long base = (long)nG * 256 + kt * 32 + koffB;
    Bf[kt].u[0] = *(const u32x4*)&UfT[base];
    Bf[kt].u[1] = *(const u32x4*)&UfT[base + 8];
    Bs[kt].u[0] = *(const u32x4*)&UsT[base];
    Bs[kt].u[1] = *(const u32x4*)&UsT[base + 8];
  }

  for (int i = tid; i < 16 * 256; i += 512) { hF[i] = 0.0f; hA[i] = 0; }
  // Stage t = 0.
  async_b128(offF[0], xf + gchunk);
  async_b128(offS[0], xs + gchunk);
  wait_async0();
  __syncthreads();

  for (int t = 0; t < SEQ; ++t) {
    const int buf = t & 1;
    if (t + 1 < SEQ) {                          // overlap DMA with both GEMMs
      async_b128(offF[buf ^ 1], xf + gchunk + (size_t)(t + 1) * 256);
      async_b128(offS[buf ^ 1], xs + gchunk + (size_t)(t + 1) * 256);
    }
    if (t + 2 < SEQ) {                          // warm L2 for the next DMA
      __builtin_prefetch(xf + gchunk + (size_t)(t + 2) * 256, 0, 1);
      __builtin_prefetch(xs + gchunk + (size_t)(t + 2) * 256, 0, 1);
    }

    // ---- GEMM1: h @ U_f (this wave's 16 columns) ----
    v8f acc = {};
    #pragma unroll
    for (int kt = 0; kt < 8; ++kt) {
      Frag a;
      a.u[0] = *(const u32x4*)&hA[laneM * 256 + kt * 32 + koffA];
      a.u[1] = *(const u32x4*)&hA[laneM * 256 + kt * 32 + koffA + 8];
      acc = wmma_bf16(a, Bf[kt], acc);
    }

    float fv[8], hv[8];
    long  idx[8];
    #pragma unroll
    for (int r = 0; r < 8; ++r) {
      const int Ml = (lane < 16) ? r : r + 8;
      idx[r] = ((long)(b0 + Ml) * SEQ + t) * 256 + nG;
      const float pre = acc[r] + bf2f(xfL[buf][Ml * 256 + nG]);
      fv[r] = 1.0f / (1.0f + __expf(-pre));     // sigmoid
      hv[r] = hF[Ml * 256 + nG];
      gA[Ml * 256 + nG] = f2bf(fv[r] * hv[r]);
    }
    __syncthreads();                            // gA complete for all waves

    // ---- GEMM2: (f*h) @ U_s ----
    v8f acc2 = {};
    #pragma unroll
    for (int kt = 0; kt < 8; ++kt) {
      Frag a;
      a.u[0] = *(const u32x4*)&gA[laneM * 256 + kt * 32 + koffA];
      a.u[1] = *(const u32x4*)&gA[laneM * 256 + kt * 32 + koffA + 8];
      acc2 = wmma_bf16(a, Bs[kt], acc2);
    }

    #pragma unroll
    for (int r = 0; r < 8; ++r) {
      const int Ml = (lane < 16) ? r : r + 8;
      const float st = tanhf(acc2[r] + bf2f(xsL[buf][Ml * 256 + nG]));
      const float hn = (1.0f - fv[r]) * hv[r] + fv[r] * st;
      hF[Ml * 256 + nG] = hn;
      hA[Ml * 256 + nG] = f2bf(hn);
      out[idx[r]] = hn;                         // (B,T,U) f32
    }
    wait_async0();                              // t+1 staging landed in LDS
    __syncthreads();                            // ... and visible to all waves
  }
}

// ---------------------------------------------------------------------------
extern "C" void kernel_launch(void* const* d_in, const int* in_sizes, int n_in,
                              void* d_out, int out_size, void* d_ws, size_t ws_size,
                              hipStream_t stream) {
  const float* inputs = (const float*)d_in[0];
  const float* W_f    = (const float*)d_in[1];
  const float* U_f    = (const float*)d_in[2];
  const float* b_f    = (const float*)d_in[3];
  const float* W_s    = (const float*)d_in[4];
  const float* U_s    = (const float*)d_in[5];
  const float* b_s    = (const float*)d_in[6];
  float* out = (float*)d_out;

  char* ws = (char*)d_ws;
  unsigned short* WfT = (unsigned short*)(ws);
  unsigned short* UfT = (unsigned short*)(ws + 131072);
  unsigned short* WsT = (unsigned short*)(ws + 262144);
  unsigned short* UsT = (unsigned short*)(ws + 393216);
  unsigned short* xf  = (unsigned short*)(ws + 524288);
  unsigned short* xs  = (unsigned short*)(ws + 524288 + (size_t)ROWS * 256 * 2);

  hipLaunchKernelGGL(k_wconv, dim3(256), dim3(256), 0, stream,
                     W_f, U_f, W_s, U_s, WfT, UfT, WsT, UsT);
  hipLaunchKernelGGL(k_xproj, dim3(ROWS / 64), dim3(256), 0, stream,
                     inputs, WfT, WsT, b_f, b_s, xf, xs);
  hipLaunchKernelGGL(k_scan, dim3(BATCH / 16), dim3(512), 0, stream,
                     xf, xs, UfT, UsT, out);
}
